// MHAtt_8589935010
// MI455X (gfx1250) — matrix operand order repaired and checked by reading, not compiled, not run on hardware
//
#include <hip/hip_runtime.h>

typedef __bf16 bf16_t;
typedef __attribute__((ext_vector_type(16))) __bf16 v16bf;
typedef __attribute__((ext_vector_type(8)))  __bf16 v8bf;
typedef __attribute__((ext_vector_type(8)))  float  v8f;

typedef unsigned int u32x4 __attribute__((ext_vector_type(4)));
typedef int          i32x4 __attribute__((ext_vector_type(4)));
typedef int          i32x8 __attribute__((ext_vector_type(8)));

#define S_LEN 1024
#define HDIM  1024
#define NHEAD 16
#define DHEAD 64

#if defined(__has_builtin)
#  if __has_builtin(__builtin_amdgcn_tensor_load_to_lds)
#    define HAVE_TDM 1
#  else
#    define HAVE_TDM 0
#  endif
#else
#  define HAVE_TDM 0
#endif

struct __align__(8) BF4 { bf16_t v[4]; };

static __device__ inline BF4 ld4(const float* p) {
  float4 t = *(const float4*)p;
  BF4 r;
  r.v[0] = (bf16_t)t.x; r.v[1] = (bf16_t)t.y;
  r.v[2] = (bf16_t)t.z; r.v[3] = (bf16_t)t.w;
  return r;
}
static __device__ inline BF4 ld4(const bf16_t* p) { return *(const BF4*)p; }

static __device__ inline v8f wmma_bf16(v16bf a, v16bf b, v8f c) {
  // D = A(16x32 bf16) * B(32x16 bf16) + C(16x16 f32)
  return __builtin_amdgcn_wmma_f32_16x16x32_bf16(
      /*neg_a=*/false, a, /*neg_b=*/false, b,
      /*c_mod=*/(short)0, c, /*reuse_a=*/false, /*reuse_b=*/false);
}

// LDS byte offset of a __shared__ pointer (ptrtoint in addrspace(3)).
static __device__ inline unsigned lds_off_u32(const void* p) {
  return (unsigned)(unsigned long long)
      (__attribute__((address_space(3))) const void*)p;
}

// Max-reduce across the 16-lane row group using DPP16 (no LDS traffic).
static __device__ inline float dpp_row_max16(float x) {
  int xi;
  xi = __builtin_amdgcn_update_dpp(0, __float_as_int(x), 0xB1, 0xF, 0xF, true);
  x = fmaxf(x, __int_as_float(xi));
  xi = __builtin_amdgcn_update_dpp(0, __float_as_int(x), 0x4E, 0xF, 0xF, true);
  x = fmaxf(x, __int_as_float(xi));
  xi = __builtin_amdgcn_update_dpp(0, __float_as_int(x), 0x141, 0xF, 0xF, true);
  x = fmaxf(x, __int_as_float(xi));
  xi = __builtin_amdgcn_update_dpp(0, __float_as_int(x), 0x140, 0xF, 0xF, true);
  x = fmaxf(x, __int_as_float(xi));
  return x;
}

// A-matrix fragment (16x32, row-major source): lane holds row = lane%16,
// K chunks [8*(lane/16) .. +7] and [+16 .. +23] (ISA 16-bit A layout).
static __device__ inline v16bf frag_a(const bf16_t* base, int stride) {
  int lane = threadIdx.x & 31;
  const bf16_t* p = base + (lane & 15) * stride + ((lane >> 4) << 3);
  v8bf c0 = *(const v8bf*)(p);
  v8bf c1 = *(const v8bf*)(p + 16);
  v16bf r;
#pragma unroll
  for (int j = 0; j < 8; ++j) { r[j] = c0[j]; r[j + 8] = c1[j]; }
  return r;
}

// B-matrix fragment (32x16). Source stored as Bt[n][k] row-major.
// Lane holds column n = lane%16, K chunks [8*(lane/16) .. +7] and [+16 .. +23].
static __device__ inline v16bf frag_b(const bf16_t* base, int stride) {
  int lane = threadIdx.x & 31;
  const bf16_t* p = base + (lane & 15) * stride + ((lane >> 4) << 3);
  v8bf c0 = *(const v8bf*)(p);
  v8bf c1 = *(const v8bf*)(p + 16);
  v16bf r;
#pragma unroll
  for (int j = 0; j < 8; ++j) { r[j] = c0[j]; r[j + 8] = c1[j]; }
  return r;
}

// ---------------------------------------------------------------------------
// GEMM: Out = A[M x 1024] * W[1024 x 1024]^T + bias.  M = 8192.
// HEADSPLIT: store bf16 into [B, NH, S, DH]; else store fp32 row-major.
// ---------------------------------------------------------------------------
template <typename TIN, bool HEADSPLIT>
__global__ __launch_bounds__(256, 2)
void gemm_bias_kernel(const TIN* __restrict__ A, const float* __restrict__ W,
                      const float* __restrict__ bias, void* __restrict__ OutV) {
  constexpr int BM = 128, BN = 64, BK = 32, LDA = BK + 8, LDB = BK + 8;
  __shared__ __align__(16) bf16_t As[BM * LDA];
  __shared__ __align__(16) bf16_t Bs[BN * LDB];

  const int m0 = blockIdx.x * BM;
  const int n0 = blockIdx.y * BN;
  const int t = threadIdx.x;
  const int wave = t >> 5;
  const int wm = wave & 3;   // 4 waves along M (32 rows each)
  const int wn = wave >> 2;  // 2 waves along N (32 cols each)

  v8f acc[2][2] = {};

  for (int k0 = 0; k0 < HDIM; k0 += BK) {
    __syncthreads();
#pragma unroll
    for (int i = 0; i < 4; ++i) {  // 128x32 A tile, cvt to bf16
      int id = i * 256 + t;
      int r = id >> 3, c = (id & 7) << 2;
      *(BF4*)(As + r * LDA + c) = ld4(A + (size_t)(m0 + r) * HDIM + k0 + c);
    }
#pragma unroll
    for (int i = 0; i < 2; ++i) {  // 64x32 W tile (row-major [n][k])
      int id = i * 256 + t;
      int r = id >> 3, c = (id & 7) << 2;
      *(BF4*)(Bs + r * LDB + c) = ld4(W + (size_t)(n0 + r) * HDIM + k0 + c);
    }
    __syncthreads();

    v16bf a0 = frag_a(As + (wm * 32 + 0) * LDA, LDA);
    v16bf a1 = frag_a(As + (wm * 32 + 16) * LDA, LDA);
    v16bf b0 = frag_b(Bs + (wn * 32 + 0) * LDB, LDB);
    v16bf b1 = frag_b(Bs + (wn * 32 + 16) * LDB, LDB);
    acc[0][0] = wmma_bf16(a0, b0, acc[0][0]);
    acc[0][1] = wmma_bf16(a0, b1, acc[0][1]);
    acc[1][0] = wmma_bf16(a1, b0, acc[1][0]);
    acc[1][1] = wmma_bf16(a1, b1, acc[1][1]);
  }

  const int lane = t & 31;
  const int col = lane & 15;
  const int rb = (lane >> 4) << 3;
#pragma unroll
  for (int sm = 0; sm < 2; ++sm) {
#pragma unroll
    for (int sn = 0; sn < 2; ++sn) {
      int n = n0 + wn * 32 + sn * 16 + col;
      float bv = bias[n];
#pragma unroll
      for (int i = 0; i < 8; ++i) {
        int m = m0 + wm * 32 + sm * 16 + rb + i;
        float val = acc[sm][sn][i] + bv;
        if (HEADSPLIT) {
          int bb = m >> 10;            // S == 1024
          int s = m & 1023;
          int head = n >> 6;           // DH == 64
          int d = n & 63;
          bf16_t* Out = (bf16_t*)OutV;
          Out[(((size_t)bb * NHEAD + head) * S_LEN + s) * DHEAD + d] = (bf16_t)val;
        } else {
          float* Out = (float*)OutV;
          Out[(size_t)m * HDIM + n] = val;
        }
      }
    }
  }
}

// ---------------------------------------------------------------------------
// Flash attention: one WG = one (b, head) x 64 query rows, 4 waves x 16 rows.
// K tile staged via Tensor Data Mover (tensor_load_to_lds + s_wait_tensorcnt);
// V tile staged transposed by the waves. Row max via DPP16, row sum via
// ones-matrix WMMA.
// ---------------------------------------------------------------------------
__global__ __launch_bounds__(128, 2)
void flash_attn_kernel(const bf16_t* __restrict__ Qh, const bf16_t* __restrict__ Kh,
                       const bf16_t* __restrict__ Vh,
                       const unsigned char* __restrict__ mask,
                       bf16_t* __restrict__ Ctx) {
  constexpr int LKS = DHEAD + 8;  // K tile stride: [32 keys][64+8 d] (144 B rows)
  constexpr int LVT = 32 + 8;     // V^T tile stride: [64 d][32+8 keys]
  __shared__ __align__(16) bf16_t Ks[32 * LKS];
  __shared__ __align__(16) bf16_t Vt[DHEAD * LVT];
  __shared__ __align__(16) bf16_t Pb[4 * 16 * 32];  // per-wave P C->A relayout

  const int bh = blockIdx.y;      // b*NH + head
  const int b = bh >> 4;
  const int head = bh & 15;
  const int wave = threadIdx.x >> 5;
  const int lane = threadIdx.x & 31;
  const int q0 = blockIdx.x * 64 + wave * 16;

  const bf16_t* Qb = Qh + (size_t)bh * S_LEN * DHEAD;
  const bf16_t* Kb = Kh + (size_t)bh * S_LEN * DHEAD;
  const bf16_t* Vb = Vh + (size_t)bh * S_LEN * DHEAD;
  const unsigned char* mrow = mask + (size_t)b * S_LEN;

  // Q fragments for this wave's 16 rows (d 0..31 and 32..63), A-layout.
  v16bf qf0, qf1;
  {
    const bf16_t* p = Qb + (size_t)(q0 + (lane & 15)) * DHEAD + ((lane >> 4) << 3);
    v8bf c0 = *(const v8bf*)(p);
    v8bf c1 = *(const v8bf*)(p + 16);
    v8bf c2 = *(const v8bf*)(p + 32);
    v8bf c3 = *(const v8bf*)(p + 48);
#pragma unroll
    for (int j = 0; j < 8; ++j) {
      qf0[j] = c0[j]; qf0[j + 8] = c1[j];
      qf1[j] = c2[j]; qf1[j + 8] = c3[j];
    }
  }

  v16bf onesb;  // B-fragment of all-ones: rowsum(P) = P x Ones via WMMA
#pragma unroll
  for (int j = 0; j < 16; ++j) onesb[j] = (bf16_t)1.0f;

#if HAVE_TDM
  const unsigned ks_lds = lds_off_u32(Ks);
#endif

  v8f o[4] = {};          // 16 rows x 64 d accumulator (4 C tiles)
  float mst[8], lst[8];   // per-row running max / sum (rows split per lane half)
#pragma unroll
  for (int i = 0; i < 8; ++i) { mst[i] = -1e30f; lst[i] = 0.f; }

  for (int kb = 0; kb < S_LEN; kb += 32) {
    __syncthreads();
    if (kb + 32 < S_LEN) {  // prefetch next K/V tiles (global_prefetch_b8)
      __builtin_prefetch(Kb + (size_t)(kb + 32) * DHEAD + ((threadIdx.x & 31) << 6), 0, 1);
      __builtin_prefetch(Vb + (size_t)(kb + 32) * DHEAD + ((threadIdx.x & 31) << 6), 0, 1);
    }

#if HAVE_TDM
    if (wave == 0) {
      // D# groups per CDNA5 ISA 8.3/8.4: 2D tensor 1024x64 bf16, tile 32x64,
      // LDS pad: 4 DWORDs after every 32 DWORDs -> 144 B row stride (== LKS).
      unsigned long long ga =
          (unsigned long long)(const void*)(Kb + (size_t)kb * DHEAD);
      u32x4 g0;
      g0[0] = 1u;                                   // count=1, user descriptor
      g0[1] = ks_lds;                               // lds_addr (bytes)
      g0[2] = (unsigned)(ga & 0xffffffffu);         // global_addr[31:0]
      g0[3] = (unsigned)(ga >> 32) | 0x80000000u;   // global_addr[56:32] | type=2
      i32x8 g1;
      g1[0] = 0x07110000;          // data_size=2B, pad_en, pad_int=32dw, pad_amt=4dw
      g1[1] = (int)(64u << 16);    // tensor_dim0 = 64 (low 16 bits)
      g1[2] = (int)(1024u << 16);  // tensor_dim1 = 1024 (low 16 bits)
      g1[3] = (int)(64u << 16);    // tile_dim0 = 64
      g1[4] = 32;                  // tile_dim1 = 32
      g1[5] = 64;                  // tensor_dim0_stride = 64 elems
      g1[6] = 0;
      g1[7] = 0;
      i32x4 g2 = {0, 0, 0, 0};
      i32x4 g3 = {0, 0, 0, 0};
#if __clang_major__ >= 23
      i32x8 g4 = {0, 0, 0, 0, 0, 0, 0, 0};
      __builtin_amdgcn_tensor_load_to_lds(g0, g1, g2, g3, g4, 0);
#else
      __builtin_amdgcn_tensor_load_to_lds(g0, g1, g2, g3, 0);
#endif
    }
#else
#pragma unroll
    for (int i = 0; i < 2; ++i) {  // stage K tile [32][64] row-major
      int id = i * 128 + threadIdx.x;
      int r = id >> 3, c = (id & 7) << 3;
      *(v8bf*)(Ks + r * LKS + c) = *(const v8bf*)(Kb + (size_t)(kb + r) * DHEAD + c);
    }
#endif

#pragma unroll
    for (int i = 0; i < 4; ++i) {  // stage V tile transposed: Vt[d][key]
      int id = i * 128 + threadIdx.x;
      int r = id >> 4, c = (id & 15) << 2;
      BF4 vv = ld4(Vb + (size_t)(kb + r) * DHEAD + c);
#pragma unroll
      for (int j = 0; j < 4; ++j) Vt[(c + j) * LVT + r] = vv.v[j];
    }

#if HAVE_TDM
    if (wave == 0) {
#if __has_builtin(__builtin_amdgcn_s_wait_tensorcnt)
      __builtin_amdgcn_s_wait_tensorcnt(0);
#else
      asm volatile("s_wait_tensorcnt 0x0" ::: "memory");
#endif
    }
#endif
    __syncthreads();

    // scores: S[q, key] = Q . K^T  (B-frags read K rows contiguously)
    v8f s0 = {}, s1 = {};
    {
      v16bf k00 = frag_b(Ks + 0 * LKS + 0, LKS);
      v16bf k01 = frag_b(Ks + 0 * LKS + 32, LKS);
      v16bf k10 = frag_b(Ks + 16 * LKS + 0, LKS);
      v16bf k11 = frag_b(Ks + 16 * LKS + 32, LKS);
      s0 = wmma_bf16(qf0, k00, s0);
      s0 = wmma_bf16(qf1, k01, s0);
      s1 = wmma_bf16(qf0, k10, s1);
      s1 = wmma_bf16(qf1, k11, s1);
    }

    const bool msk0 = mrow[kb + (lane & 15)] != 0;
    const bool msk1 = mrow[kb + 16 + (lane & 15)] != 0;

    float alpha[8], e0v[8], e1v[8];
#pragma unroll
    for (int i = 0; i < 8; ++i) {
      float a = msk0 ? -1e9f : s0[i] * 0.125f;   // 1/sqrt(64)
      float c = msk1 ? -1e9f : s1[i] * 0.125f;
      float mx = dpp_row_max16(fmaxf(a, c));     // DPP16 reduction, no LDS
      float mn = fmaxf(mst[i], mx);
      alpha[i] = __expf(mst[i] - mn);
      mst[i] = mn;
      e0v[i] = __expf(a - mn);
      e1v[i] = __expf(c - mn);
#pragma unroll
      for (int tt = 0; tt < 4; ++tt) o[tt][i] *= alpha[i];
    }

    // P: C-layout -> LDS -> A-layout fragment (per-wave buffer, wave-ordered DS)
    bf16_t* pb = Pb + wave * (16 * 32);
    {
      int c0i = lane & 15;
      int r0 = (lane >> 4) << 3;
#pragma unroll
      for (int i = 0; i < 8; ++i) {
        pb[(r0 + i) * 32 + c0i] = (bf16_t)e0v[i];
        pb[(r0 + i) * 32 + 16 + c0i] = (bf16_t)e1v[i];
      }
    }
    asm volatile("" ::: "memory");
    v16bf pf = frag_a(pb, 32);

    // row sums of P via WMMA with all-ones B matrix: srow[i] = sum_k P[row_i,k]
    v8f srow = {};
    srow = wmma_bf16(pf, onesb, srow);
#pragma unroll
    for (int i = 0; i < 8; ++i) lst[i] = lst[i] * alpha[i] + srow[i];

#pragma unroll
    for (int tt = 0; tt < 4; ++tt) {
      v16bf vf = frag_b(Vt + (tt * 16) * LVT, LVT);  // B-frag from V^T rows
      o[tt] = wmma_bf16(pf, vf, o[tt]);
    }
  }

  // epilogue: normalize and store ctx merged [B, S, H] bf16
  {
    int cl = lane & 15;
    int r0 = (lane >> 4) << 3;
#pragma unroll
    for (int tt = 0; tt < 4; ++tt) {
      int d = tt * 16 + cl;
#pragma unroll
      for (int i = 0; i < 8; ++i) {
        int qrow = q0 + r0 + i;
        float val = o[tt][i] / lst[i];
        Ctx[((size_t)b * S_LEN + qrow) * HDIM + head * DHEAD + d] = (bf16_t)val;
      }
    }
  }
}

// ---------------------------------------------------------------------------
extern "C" void kernel_launch(void* const* d_in, const int* in_sizes, int n_in,
                              void* d_out, int out_size, void* d_ws, size_t ws_size,
                              hipStream_t stream) {
  (void)in_sizes; (void)n_in; (void)out_size; (void)ws_size;
  const float* v = (const float*)d_in[0];
  const float* k = (const float*)d_in[1];
  const float* q = (const float*)d_in[2];
  const unsigned char* mask = (const unsigned char*)d_in[3];  // bool mask
  const float* Wq = (const float*)d_in[4];
  const float* bq = (const float*)d_in[5];
  const float* Wk = (const float*)d_in[6];
  const float* bk = (const float*)d_in[7];
  const float* Wv = (const float*)d_in[8];
  const float* bv = (const float*)d_in[9];
  const float* Wm = (const float*)d_in[10];
  const float* bm = (const float*)d_in[11];

  const size_t per = (size_t)8 * NHEAD * S_LEN * DHEAD;  // 8 Mi elems (16 MB bf16)
  bf16_t* Qh = (bf16_t*)d_ws;
  bf16_t* Kh = Qh + per;
  bf16_t* Vh = Kh + per;
  bf16_t* Ctx = Vh + per;  // [B, S, H] bf16 (total 64 MB workspace)

  dim3 gg(64, 16), gb(256);   // M=8192 / 128, N=1024 / 64
  gemm_bias_kernel<float, true><<<gg, gb, 0, stream>>>(q, Wq, bq, (void*)Qh);
  gemm_bias_kernel<float, true><<<gg, gb, 0, stream>>>(k, Wk, bk, (void*)Kh);
  gemm_bias_kernel<float, true><<<gg, gb, 0, stream>>>(v, Wv, bv, (void*)Vh);

  dim3 ga(S_LEN / 64, 8 * NHEAD), ba(128);
  flash_attn_kernel<<<ga, ba, 0, stream>>>(Qh, Kh, Vh, mask, Ctx);

  gemm_bias_kernel<bf16_t, false><<<gg, gb, 0, stream>>>(Ctx, Wm, bm, d_out);
}
